// QLlamaMLP_25572235281042
// MI455X (gfx1250) — matrix operand order repaired and checked by reading, not compiled, not run on hardware
//
#include <hip/hip_runtime.h>
#include <cstdint>

// Problem constants (from reference)
#define T_TOK  2048
#define HID    4096
#define INTER_ 11008
#define SELN   128

#define KSTEP  64                 // K halves staged per LDS buffer
#define ROWP   72                 // padded LDS row pitch in halves (144 B, 16B-aligned)

typedef __attribute__((ext_vector_type(16))) _Float16 v16h;
typedef __attribute__((ext_vector_type(8)))  _Float16 v8h;
typedef __attribute__((ext_vector_type(8)))  float    v8f;

union Frag16 { v16h v; v8h h[2]; };

// ---------------------------------------------------------------------------
// CDNA5 async global -> LDS copy (tracked by ASYNCcnt), 16 bytes per lane.
// LDS address: flat address truncated to 32 bits.
// ---------------------------------------------------------------------------
__device__ __forceinline__ void async_ld16(const void* lds_ptr, const void* gptr) {
    unsigned loff = (unsigned)(uintptr_t)lds_ptr;
    asm volatile("global_load_async_to_lds_b128 %0, %1, off"
                 :: "v"(loff), "v"(gptr) : "memory");
}
__device__ __forceinline__ void wait_async0() {
    asm volatile("s_wait_asynccnt 0" ::: "memory");
}

__device__ __forceinline__ v8f wmma16(const Frag16& a, const Frag16& b, v8f c) {
    return __builtin_amdgcn_wmma_f32_16x16x32_f16(false, a.v, false, b.v,
                                                  (short)0, c, false, false);
}

// ---------------------------------------------------------------------------
// Quant-dequant helpers (NVFP4: E2M1 elements, E4M3 block scale)
// ---------------------------------------------------------------------------
__device__ __forceinline__ float qdq_e4m3(float x) {
    float ax = fabsf(x);
    ax = fminf(ax, 448.0f);
    unsigned bits = __float_as_uint(ax);
    int e = (int)(bits >> 23) - 127;
    float step = (ax >= 0.015625f)
                     ? __uint_as_float((unsigned)(e - 3 + 127) << 23)
                     : 0x1p-9f;
    float q = rintf(ax / step) * step;
    q = fminf(q, 448.0f);
    return copysignf(q, x);
}

__device__ __forceinline__ float qdq_e2m1(float v) {
    float a = fminf(fabsf(v), 6.0f);
    float q;
    if      (a < 0.25f) q = 0.0f;
    else if (a < 0.75f) q = 0.5f;
    else if (a < 1.25f) q = 1.0f;
    else if (a < 1.75f) q = 1.5f;
    else if (a < 2.5f)  q = 2.0f;
    else if (a < 3.5f)  q = 3.0f;
    else if (a < 5.0f)  q = 4.0f;
    else                q = 6.0f;
    return copysignf(q, v);
}

// ---------------------------------------------------------------------------
// Gathered amax over non-outlier (reordered) columns
// ---------------------------------------------------------------------------
__global__ __launch_bounds__(256) void amax_gather_kernel(
    const float* __restrict__ src, const long long* __restrict__ idx,
    int rows, int cols, int sel, float* __restrict__ out_amax) {
    __shared__ float red[256];
    const int ncq = cols - sel;
    const long long total = (long long)rows * ncq;
    float m = 0.0f;
    for (long long i = (long long)blockIdx.x * blockDim.x + threadIdx.x;
         i < total; i += (long long)gridDim.x * blockDim.x) {
        int r = (int)(i / ncq);
        int j = (int)(i % ncq) + sel;
        float v = src[(long long)r * cols + idx[j]];
        m = fmaxf(m, fabsf(v));
    }
    red[threadIdx.x] = m;
    __syncthreads();
    for (int s = 128; s > 0; s >>= 1) {
        if ((int)threadIdx.x < s) red[threadIdx.x] = fmaxf(red[threadIdx.x], red[threadIdx.x + s]);
        __syncthreads();
    }
    if (threadIdx.x == 0)
        atomicMax((unsigned int*)out_amax, __float_as_uint(red[0]));
}

// ---------------------------------------------------------------------------
// Reorder + NVFP4 fake quant-dequant -> FP16 operand buffer
// ---------------------------------------------------------------------------
__global__ __launch_bounds__(256) void reorder_quant_kernel(
    const float* __restrict__ src, const long long* __restrict__ idx,
    int rows, int cols, int sel, const float* __restrict__ amax_ptr,
    _Float16* __restrict__ dst) {
    const int bpr = cols >> 4;
    const long long nblk = (long long)rows * bpr;
    const long long i = (long long)blockIdx.x * blockDim.x + threadIdx.x;
    if (i >= nblk) return;
    const int r = (int)(i / bpr);
    const int b = (int)(i % bpr);
    const long long rowoff = (long long)r * cols;
    const int c0 = b << 4;

    float v[16];
#pragma unroll
    for (int k = 0; k < 16; ++k) v[k] = src[rowoff + idx[c0 + k]];

    _Float16 o[16];
    if (c0 < sel) {
#pragma unroll
        for (int k = 0; k < 16; ++k) o[k] = (_Float16)v[k];
    } else {
        const float s = fmaxf(amax_ptr[0] * (1.0f / (448.0f * 6.0f)), 1e-12f);
        const float inv_s = 1.0f / s;
        float bamax = 0.0f;
#pragma unroll
        for (int k = 0; k < 16; ++k) bamax = fmaxf(bamax, fabsf(v[k] * inv_s));
        const float bscale = fmaxf(qdq_e4m3(bamax * (1.0f / 6.0f)), 0x1p-10f);
        const float inv_b = 1.0f / bscale;
        const float back = bscale * s;
#pragma unroll
        for (int k = 0; k < 16; ++k)
            o[k] = (_Float16)(qdq_e2m1(v[k] * inv_s * inv_b) * back);
    }
    v8h* p = (v8h*)(dst + rowoff + c0);
    v8h lo, hi;
#pragma unroll
    for (int k = 0; k < 8; ++k) { lo[k] = o[k]; hi[k] = o[k + 8]; }
    p[0] = lo;
    p[1] = hi;
}

// ---------------------------------------------------------------------------
// GEMM1 + SwiGLU: h = silu(Xq@Gqt) * (Xq@Uqt)
// WG tile 64M x 64N, K staged 64-wide in LDS (A + Wg + Wu), double-buffered
// via async global->LDS. 8 waves as 2(M) x 4(N); each wave: 2 M-subtiles x 1
// N-subtile with dual (gate,up) accumulators -> 4 WMMAs per 32-K inner step.
// ---------------------------------------------------------------------------
__global__ __launch_bounds__(256) void gemm1_swiglu_wmma_kernel(
    const _Float16* __restrict__ A, const _Float16* __restrict__ Wg,
    const _Float16* __restrict__ Wu, float* __restrict__ H, int K, int N) {
    __shared__ __align__(16) _Float16 sA[2][64 * ROWP];
    __shared__ __align__(16) _Float16 sG[2][64 * ROWP];
    __shared__ __align__(16) _Float16 sU[2][64 * ROWP];

    const int tid  = threadIdx.x;
    const int lane = tid & 31;
    const int wid  = tid >> 5;
    const int waveM = wid >> 2;      // 0..1
    const int waveN = wid & 3;       // 0..3
    const int ml = lane & 15;
    const int hi = lane >> 4;

    const int mBase = blockIdx.y * 64;
    const int nBase = blockIdx.x * 64;

    // 64 rows x 128B per tile -> 512 x 16B chunks -> 2 per thread per tile
    auto issueTile = [&](int buf, int k0) {
#pragma unroll
        for (int c = 0; c < 2; ++c) {
            const int ch  = c * 256 + tid;       // 0..511
            const int row = ch >> 3;
            const int seg = (ch & 7) * 8;        // half offset within K slice
            async_ld16(&sA[buf][row * ROWP + seg],
                       A  + (long long)(mBase + row) * K + k0 + seg);
            async_ld16(&sG[buf][row * ROWP + seg],
                       Wg + (long long)(nBase + row) * K + k0 + seg);
            async_ld16(&sU[buf][row * ROWP + seg],
                       Wu + (long long)(nBase + row) * K + k0 + seg);
        }
    };

    v8f accg[2] = {};
    v8f accu[2] = {};

    const int nk = K / KSTEP;
    issueTile(0, 0);
    for (int ks = 0; ks < nk; ++ks) {
        wait_async0();
        __syncthreads();
        if (ks + 1 < nk) issueTile((ks + 1) & 1, (ks + 1) * KSTEP);
        const int buf = ks & 1;
#pragma unroll
        for (int kk = 0; kk < KSTEP; kk += 32) {
            Frag16 af[2], bg, bu;
#pragma unroll
            for (int i = 0; i < 2; ++i) {
                const _Float16* p = &sA[buf][(waveM * 32 + i * 16 + ml) * ROWP + kk + hi * 16];
                af[i].h[0] = *(const v8h*)p;
                af[i].h[1] = *(const v8h*)(p + 8);
            }
            const _Float16* pg = &sG[buf][(waveN * 16 + ml) * ROWP + kk + hi * 16];
            bg.h[0] = *(const v8h*)pg;
            bg.h[1] = *(const v8h*)(pg + 8);
            const _Float16* pu = &sU[buf][(waveN * 16 + ml) * ROWP + kk + hi * 16];
            bu.h[0] = *(const v8h*)pu;
            bu.h[1] = *(const v8h*)(pu + 8);
#pragma unroll
            for (int i = 0; i < 2; ++i) {
                accg[i] = wmma16(af[i], bg, accg[i]);
                accu[i] = wmma16(af[i], bu, accu[i]);
            }
        }
    }

#pragma unroll
    for (int i = 0; i < 2; ++i) {
#pragma unroll
        for (int r = 0; r < 8; ++r) {
            const int M  = mBase + waveM * 32 + i * 16 + hi * 8 + r;
            const int Nc = nBase + waveN * 16 + ml;
            const float g = accg[i][r];
            const float u = accu[i][r];
            H[(long long)M * N + Nc] = (g / (1.0f + expf(-g))) * u;
        }
    }
}

// ---------------------------------------------------------------------------
// GEMM2: out = Hq @ Dqt. WG tile 128M x 64N, async-LDS double buffered.
// 8 waves as 4(M) x 2(N); each wave 32x32 = 2x2 subtiles -> 8 WMMAs / K-step.
// ---------------------------------------------------------------------------
__global__ __launch_bounds__(256) void gemm2_wmma_kernel(
    const _Float16* __restrict__ A, const _Float16* __restrict__ W,
    float* __restrict__ out, int K, int N) {
    __shared__ __align__(16) _Float16 sA[2][128 * ROWP];
    __shared__ __align__(16) _Float16 sB[2][64 * ROWP];

    const int tid  = threadIdx.x;
    const int lane = tid & 31;
    const int wid  = tid >> 5;
    const int waveM = wid >> 1;      // 0..3
    const int waveN = wid & 1;       // 0..1
    const int ml = lane & 15;
    const int hi = lane >> 4;

    const int mBase = blockIdx.y * 128;
    const int nBase = blockIdx.x * 64;

    auto issueTile = [&](int buf, int k0) {
        // A: 128 rows x 8 chunks = 1024 chunks -> 4 per thread
#pragma unroll
        for (int c = 0; c < 4; ++c) {
            const int ch  = c * 256 + tid;       // 0..1023
            const int row = ch >> 3;
            const int seg = (ch & 7) * 8;
            async_ld16(&sA[buf][row * ROWP + seg],
                       A + (long long)(mBase + row) * K + k0 + seg);
        }
        // B: 64 rows x 8 chunks = 512 chunks -> 2 per thread
#pragma unroll
        for (int c = 0; c < 2; ++c) {
            const int ch  = c * 256 + tid;       // 0..511
            const int row = ch >> 3;
            const int seg = (ch & 7) * 8;
            async_ld16(&sB[buf][row * ROWP + seg],
                       W + (long long)(nBase + row) * K + k0 + seg);
        }
    };

    v8f acc[2][2] = {};

    const int nk = K / KSTEP;
    issueTile(0, 0);
    for (int ks = 0; ks < nk; ++ks) {
        wait_async0();
        __syncthreads();
        if (ks + 1 < nk) issueTile((ks + 1) & 1, (ks + 1) * KSTEP);
        const int buf = ks & 1;
#pragma unroll
        for (int kk = 0; kk < KSTEP; kk += 32) {
            Frag16 af[2], bf[2];
#pragma unroll
            for (int i = 0; i < 2; ++i) {
                const _Float16* p = &sA[buf][(waveM * 32 + i * 16 + ml) * ROWP + kk + hi * 16];
                af[i].h[0] = *(const v8h*)p;
                af[i].h[1] = *(const v8h*)(p + 8);
                const _Float16* q = &sB[buf][(waveN * 32 + i * 16 + ml) * ROWP + kk + hi * 16];
                bf[i].h[0] = *(const v8h*)q;
                bf[i].h[1] = *(const v8h*)(q + 8);
            }
#pragma unroll
            for (int i = 0; i < 2; ++i)
#pragma unroll
                for (int j = 0; j < 2; ++j)
                    acc[i][j] = wmma16(af[i], bf[j], acc[i][j]);
        }
    }

#pragma unroll
    for (int i = 0; i < 2; ++i)
#pragma unroll
        for (int j = 0; j < 2; ++j)
#pragma unroll
            for (int r = 0; r < 8; ++r) {
                const int M  = mBase + waveM * 32 + i * 16 + hi * 8 + r;
                const int Nc = nBase + waveN * 32 + j * 16 + ml;
                out[(long long)M * N + Nc] = acc[i][j][r];
            }
}

// ---------------------------------------------------------------------------
// Host side
// ---------------------------------------------------------------------------
extern "C" void kernel_launch(void* const* d_in, const int* in_sizes, int n_in,
                              void* d_out, int out_size, void* d_ws, size_t ws_size,
                              hipStream_t stream) {
    (void)in_sizes; (void)n_in; (void)out_size; (void)ws_size;

    const float*     x       = (const float*)d_in[0];
    const float*     gate_w  = (const float*)d_in[1];
    const float*     up_w    = (const float*)d_in[2];
    const float*     down_w  = (const float*)d_in[3];
    const long long* idx_in  = (const long long*)d_in[4];
    const long long* idx_mid = (const long long*)d_in[5];
    float*           out     = (float*)d_out;

    char* ws = (char*)d_ws;
    float* amax = (float*)ws;   // [0]=x [1]=gate [2]=up [3]=down [4]=h

    constexpr size_t OFF_XQ = 256;
    constexpr size_t SZ_XQ  = (size_t)T_TOK * HID * 2;
    constexpr size_t OFF_GQ = OFF_XQ + SZ_XQ;
    constexpr size_t SZ_WQ  = (size_t)INTER_ * HID * 2;
    constexpr size_t OFF_UQ = OFF_GQ + SZ_WQ;
    constexpr size_t OFF_H  = OFF_UQ + SZ_WQ;
    constexpr size_t OFF_DQ = OFF_GQ;   // reuse after GEMM1
    constexpr size_t OFF_HQ = OFF_UQ;   // reuse after GEMM1

    _Float16* xq   = (_Float16*)(ws + OFF_XQ);
    _Float16* gq   = (_Float16*)(ws + OFF_GQ);
    _Float16* uq   = (_Float16*)(ws + OFF_UQ);
    float*    hbuf = (float*)   (ws + OFF_H);
    _Float16* dq   = (_Float16*)(ws + OFF_DQ);
    _Float16* hq   = (_Float16*)(ws + OFF_HQ);

    hipMemsetAsync(ws, 0, 256, stream);

    amax_gather_kernel<<<512, 256, 0, stream>>>(x,      idx_in,  T_TOK,  HID,    SELN, amax + 0);
    amax_gather_kernel<<<1024, 256, 0, stream>>>(gate_w, idx_in,  INTER_, HID,    SELN, amax + 1);
    amax_gather_kernel<<<1024, 256, 0, stream>>>(up_w,   idx_in,  INTER_, HID,    SELN, amax + 2);
    amax_gather_kernel<<<1024, 256, 0, stream>>>(down_w, idx_mid, HID,    INTER_, SELN, amax + 3);

    {
        long long nblk = (long long)T_TOK * (HID / 16);
        reorder_quant_kernel<<<(unsigned)((nblk + 255) / 256), 256, 0, stream>>>(
            x, idx_in, T_TOK, HID, SELN, amax + 0, xq);
    }
    {
        long long nblk = (long long)INTER_ * (HID / 16);
        reorder_quant_kernel<<<(unsigned)((nblk + 255) / 256), 256, 0, stream>>>(
            gate_w, idx_in, INTER_, HID, SELN, amax + 1, gq);
        reorder_quant_kernel<<<(unsigned)((nblk + 255) / 256), 256, 0, stream>>>(
            up_w, idx_in, INTER_, HID, SELN, amax + 2, uq);
    }

    {
        dim3 grid(INTER_ / 64, T_TOK / 64);
        gemm1_swiglu_wmma_kernel<<<grid, 256, 0, stream>>>(xq, gq, uq, hbuf, HID, INTER_);
    }

    {
        long long nblk = (long long)HID * (INTER_ / 16);
        reorder_quant_kernel<<<(unsigned)((nblk + 255) / 256), 256, 0, stream>>>(
            down_w, idx_mid, HID, INTER_, SELN, amax + 3, dq);
    }

    amax_gather_kernel<<<1024, 256, 0, stream>>>(hbuf, idx_mid, T_TOK, INTER_, SELN, amax + 4);
    {
        long long nblk = (long long)T_TOK * (INTER_ / 16);
        reorder_quant_kernel<<<(unsigned)((nblk + 255) / 256), 256, 0, stream>>>(
            hbuf, idx_mid, T_TOK, INTER_, SELN, amax + 4, hq);
    }

    {
        dim3 grid(HID / 64, T_TOK / 128);
        gemm2_wmma_kernel<<<grid, 256, 0, stream>>>(hq, dq, out, INTER_, HID);
    }
}
